// GraphAwareAttentionLayer_36292473651256
// MI455X (gfx1250) — matrix operand order repaired and checked by reading, not compiled, not run on hardware
//
#include <hip/hip_runtime.h>
#include <hip/hip_bf16.h>
#include <math.h>

#define BB 8
#define NN 2048
#define DD 256

typedef __bf16 bf16_t;
typedef __attribute__((ext_vector_type(16))) __bf16 v16bf;
typedef __attribute__((ext_vector_type(8)))  __bf16 bf16x8;
typedef __attribute__((ext_vector_type(8)))  float  v8f;

__device__ __forceinline__ bf16_t f2bf(float f) {
  union { float f; unsigned u; } v; v.f = f;
  unsigned r = v.u + 0x7FFFu + ((v.u >> 16) & 1u);   // round-to-nearest-even
  unsigned short h = (unsigned short)(r >> 16);
  return __builtin_bit_cast(bf16_t, h);
}

// ---- WMMA fragment loaders (wave32 layouts per CDNA5 ISA) ----
// A (16x32 bf16, row-major source): lane L holds row M=L&15.
//   lanes 0-15:  elems 0-7 = K0..7,  elems 8-15 = K16..23
//   lanes 16-31: elems 0-7 = K8..15, elems 8-15 = K24..31
__device__ __forceinline__ v16bf load_a16(const bf16_t* base, int ld) {
  int lane = threadIdx.x & 31;
  const bf16_t* p = base + (size_t)(lane & 15) * ld + ((lane & 16) ? 8 : 0);
  bf16x8 lo = *(const bf16x8*)p;
  bf16x8 hi = *(const bf16x8*)(p + 16);
  return __builtin_shufflevector(lo, hi, 0, 1, 2, 3, 4, 5, 6, 7,
                                 8, 9, 10, 11, 12, 13, 14, 15);
}
// B (32x16 bf16) from the TRANSPOSED operand stored row-major (row n = col n of B):
//   lane L holds col N=L&15; lanes 0-15 K=0..7,16..23, lanes 16-31 K=8..15,24..31.
__device__ __forceinline__ v16bf load_b16(const bf16_t* baseT, int ld) {
  int lane = threadIdx.x & 31;
  const bf16_t* p = baseT + (size_t)(lane & 15) * ld + ((lane & 16) ? 8 : 0);
  bf16x8 lo = *(const bf16x8*)p;
  bf16x8 hi = *(const bf16x8*)(p + 16);
  return __builtin_shufflevector(lo, hi, 0, 1, 2, 3, 4, 5, 6, 7,
                                 8, 9, 10, 11, 12, 13, 14, 15);
}

__device__ __forceinline__ v8f wmma_bf16(v16bf a, v16bf b, v8f c) {
  return __builtin_amdgcn_wmma_f32_16x16x32_bf16(false, a, false, b, (short)0, c,
                                                 false, false);
}

// ------------------------- prep kernels -------------------------
__global__ void cast_bf16_kernel(const float* __restrict__ src,
                                 bf16_t* __restrict__ dst, int n) {
  int stride = gridDim.x * blockDim.x;
  for (int i = blockIdx.x * blockDim.x + threadIdx.x; i < n; i += stride)
    dst[i] = f2bf(src[i]);
}

__global__ void transpose2d_bf16_kernel(const float* __restrict__ src,
                                        bf16_t* __restrict__ dst, int R, int C) {
  int n = R * C, stride = gridDim.x * blockDim.x;
  for (int i = blockIdx.x * blockDim.x + threadIdx.x; i < n; i += stride) {
    int r = i / C, c = i % C;
    dst[(size_t)c * R + r] = f2bf(src[i]);
  }
}

__global__ void transpose_z_kernel(const float* __restrict__ Z,
                                   bf16_t* __restrict__ ZT) {
  int n = BB * NN * DD, stride = gridDim.x * blockDim.x;
  for (int i = blockIdx.x * blockDim.x + threadIdx.x; i < n; i += stride) {
    int b = i / (NN * DD), rem = i % (NN * DD);
    int nn = rem / DD, d = rem % DD;
    ZT[((size_t)b * DD + d) * NN + nn] = f2bf(Z[i]);
  }
}

// ---- projections: Qcat=[Z*Wqr | Z*Wqi], Kcat=[Z*Wkr | Z*Wki]  (bf16, K-major rows)
__global__ void __launch_bounds__(128) proj_kernel(
    const bf16_t* __restrict__ Zh, const bf16_t* __restrict__ WT,
    bf16_t* __restrict__ Qcat, bf16_t* __restrict__ Kcat) {
  int wave = threadIdx.x >> 5, lane = threadIdx.x & 31;
  int tile = blockIdx.x * 4 + wave;        // (BB*NN/16)*2*16 = 32768 tiles
  int jt = tile & 15; tile >>= 4;
  int isK = tile & 1; tile >>= 1;
  int m0 = tile * 16;                      // flat row over B*N
  const bf16_t* wt = WT + ((size_t)(isK * 2 + (jt >> 3)) * 128 + (jt & 7) * 16) * 256;
  v8f c0 = {}, c1 = {};                    // dual chains: no WMMA RAW hazard stalls
#pragma unroll
  for (int k = 0; k < 8; k += 2) {
    c0 = wmma_bf16(load_a16(Zh + (size_t)m0 * DD + k * 32, DD),
                   load_b16(wt + k * 32, DD), c0);
    c1 = wmma_bf16(load_a16(Zh + (size_t)m0 * DD + (k + 1) * 32, DD),
                   load_b16(wt + (k + 1) * 32, DD), c1);
  }
  v8f c = c0 + c1;
  bf16_t* out = isK ? Kcat : Qcat;
  int mb = (lane & 16) ? 8 : 0;
  int col = (jt >> 3) * 128 + (jt & 7) * 16 + (lane & 15);
#pragma unroll
  for (int v = 0; v < 8; ++v)
    out[(size_t)(m0 + mb + v) * 256 + col] = f2bf(c[v]);
}

// ---- LHh = bf16(L @ Z)  (per batch)
__global__ void __launch_bounds__(128) lh_kernel(
    const bf16_t* __restrict__ Lh, const bf16_t* __restrict__ ZT,
    bf16_t* __restrict__ LHh) {
  int wave = threadIdx.x >> 5, lane = threadIdx.x & 31;
  int tile = blockIdx.x * 4 + wave;        // BB*(NN/16)*(DD/16) = 16384 tiles
  int dt = tile & 15; tile >>= 4;
  int nr0 = (tile & 127) * 16;
  int b = tile >> 7;
  const bf16_t* zt = ZT + ((size_t)b * DD + dt * 16) * NN;
  const bf16_t* lb = Lh + (size_t)nr0 * NN;
  v8f c0 = {}, c1 = {};
  for (int k = 0; k < NN; k += 64) {
    c0 = wmma_bf16(load_a16(lb + k, NN), load_b16(zt + k, NN), c0);
    c1 = wmma_bf16(load_a16(lb + k + 32, NN), load_b16(zt + k + 32, NN), c1);
  }
  v8f c = c0 + c1;
  int mb = (lane & 16) ? 8 : 0, col = dt * 16 + (lane & 15);
#pragma unroll
  for (int v = 0; v < 8; ++v)
    LHh[((size_t)b * NN + nr0 + mb + v) * DD + col] = f2bf(c[v]);
}

// ---- flash attention: U = softmax(mask(QK^T/16)) @ Z  (f32 out)
__global__ void __launch_bounds__(128) attn_kernel(
    const bf16_t* __restrict__ Qcat, const bf16_t* __restrict__ Kcat,
    const bf16_t* __restrict__ ZT, const unsigned char* __restrict__ adj,
    float* __restrict__ U) {
  __shared__ bf16_t lds_p[4][16 * 32];     // per-wave C->A layout bounce
  int wave = threadIdx.x >> 5, lane = threadIdx.x & 31;
  int tile = blockIdx.x * 4 + wave;        // BB*NN/16 = 1024 tiles
  int b = tile >> 7;
  int mq0 = (tile & 127) * 16;             // query row within batch
  size_t rowflat = (size_t)b * NN + mq0;
  int mb = (lane & 16) ? 8 : 0;
  int nl = lane & 15;

  v8f zero = {};
  v8f O[16];
#pragma unroll
  for (int t = 0; t < 16; ++t) O[t] = zero;
  float mrun[8], lrun[8];
#pragma unroll
  for (int v = 0; v < 8; ++v) { mrun[v] = -1e30f; lrun[v] = 0.f; }

  v16bf qa[8];                             // hoisted: invariant across key loop
#pragma unroll
  for (int k = 0; k < 8; ++k)
    qa[k] = load_a16(Qcat + rowflat * 256 + k * 32, 256);

  bf16_t* pl = lds_p[wave];

  for (int n0 = 0; n0 < NN; n0 += 32) {
    // two 16x16 score tiles, K = 256 (packed real+imag), dual chains each
    v8f s[2];
#pragma unroll
    for (int t = 0; t < 2; ++t) {
      const bf16_t* kb = Kcat + ((size_t)b * NN + n0 + t * 16) * 256;
      v8f c0 = zero, c1 = zero;
#pragma unroll
      for (int k = 0; k < 8; k += 2) {
        c0 = wmma_bf16(qa[k], load_b16(kb + k * 32, 256), c0);
        c1 = wmma_bf16(qa[k + 1], load_b16(kb + (k + 1) * 32, 256), c1);
      }
      s[t] = c0 + c1;
    }
    // scale by 1/sqrt(256) and apply graph mask
#pragma unroll
    for (int t = 0; t < 2; ++t) {
      int nk = n0 + t * 16 + nl;
#pragma unroll
      for (int v = 0; v < 8; ++v) {
        float val = s[t][v] * 0.0625f;
        s[t][v] = adj[(size_t)(mq0 + mb + v) * NN + nk] ? val : -1e9f;
      }
    }
    // online softmax: row stats via width-16 shuffle trees (rows live per-v)
    float alpha[8];
#pragma unroll
    for (int v = 0; v < 8; ++v) {
      float rm = fmaxf(s[0][v], s[1][v]);
#pragma unroll
      for (int off = 8; off >= 1; off >>= 1) rm = fmaxf(rm, __shfl_xor(rm, off, 16));
      float mnew = fmaxf(mrun[v], rm);
      alpha[v] = __expf(mrun[v] - mnew);
      mrun[v] = mnew;
      float rs = 0.f;
#pragma unroll
      for (int t = 0; t < 2; ++t) {
        float p = __expf(s[t][v] - mnew);
        s[t][v] = p;
        rs += p;
      }
#pragma unroll
      for (int off = 8; off >= 1; off >>= 1) rs += __shfl_xor(rs, off, 16);
      lrun[v] = lrun[v] * alpha[v] + rs;
    }
#pragma unroll
    for (int t = 0; t < 16; ++t)
#pragma unroll
      for (int v = 0; v < 8; ++v) O[t][v] *= alpha[v];
    // P tile: C layout -> row-major bf16 in LDS -> A fragment
#pragma unroll
    for (int t = 0; t < 2; ++t)
#pragma unroll
      for (int v = 0; v < 8; ++v)
        pl[(mb + v) * 32 + t * 16 + nl] = f2bf(s[t][v]);
    asm volatile("s_wait_dscnt 0x0" ::: "memory");
    v16bf pa = load_a16(pl, 32);
    // O += P @ Z  over all 16 d-tiles (independent accumulators: no hazards)
#pragma unroll
    for (int t = 0; t < 16; ++t) {
      v16bf bb = load_b16(ZT + ((size_t)b * DD + t * 16) * NN + n0, NN);
      O[t] = wmma_bf16(pa, bb, O[t]);
    }
  }
#pragma unroll
  for (int v = 0; v < 8; ++v) lrun[v] = 1.f / lrun[v];
#pragma unroll
  for (int t = 0; t < 16; ++t)
#pragma unroll
    for (int v = 0; v < 8; ++v)
      U[(rowflat + mb + v) * 256 + t * 16 + nl] = O[t][v] * lrun[v];
}

// ---- fused gate MLP + final product: out = sigmoid(gelu([Z|LH]W1+b1)W2+b2) * U
__global__ void __launch_bounds__(32) mlp_kernel(
    const bf16_t* __restrict__ Zh, const bf16_t* __restrict__ LHh,
    const bf16_t* __restrict__ W1T, const bf16_t* __restrict__ W2T,
    const float* __restrict__ b1, const float* __restrict__ b2,
    const float* __restrict__ U, float* __restrict__ out) {
  __shared__ bf16_t hbuf[16 * 1024];       // h tile, bf16, 32KB of 320KB LDS
  int lane = threadIdx.x & 31;
  size_t r0 = (size_t)blockIdx.x * 16;     // flat row over B*N
  int mb = (lane & 16) ? 8 : 0;
  int nl = lane & 15;
  v8f zero = {};

  v16bf za[8], la[8];                      // hoisted A fragments (K=512)
#pragma unroll
  for (int k = 0; k < 8; ++k) {
    za[k] = load_a16(Zh  + r0 * 256 + k * 32, 256);
    la[k] = load_a16(LHh + r0 * 256 + k * 32, 256);
  }
  // phase 1: h = gelu(gate_in @ W1 + b1) -> LDS
  for (int j0 = 0; j0 < 1024; j0 += 16) {
    const bf16_t* w1 = W1T + (size_t)j0 * 512;
    v8f c0 = zero, c1 = zero;
#pragma unroll
    for (int k = 0; k < 8; k += 2) {
      c0 = wmma_bf16(za[k], load_b16(w1 + k * 32, 512), c0);
      c1 = wmma_bf16(za[k + 1], load_b16(w1 + (k + 1) * 32, 512), c1);
    }
#pragma unroll
    for (int k = 0; k < 8; k += 2) {
      c0 = wmma_bf16(la[k], load_b16(w1 + 256 + k * 32, 512), c0);
      c1 = wmma_bf16(la[k + 1], load_b16(w1 + 256 + (k + 1) * 32, 512), c1);
    }
    v8f c = c0 + c1;
    float bias = b1[j0 + nl];
#pragma unroll
    for (int v = 0; v < 8; ++v) {
      float x = c[v] + bias;
      float g = 0.5f * x * (1.f + erff(x * 0.70710678118f));  // exact GELU
      hbuf[(mb + v) * 1024 + j0 + nl] = f2bf(g);
    }
  }
  asm volatile("s_wait_dscnt 0x0" ::: "memory");
  // phase 2: gate = sigmoid(h @ W2 + b2); out = gate * U
  for (int d0 = 0; d0 < 256; d0 += 16) {
    const bf16_t* w2 = W2T + (size_t)d0 * 1024;
    v8f c0 = zero, c1 = zero;
    for (int k = 0; k < 1024; k += 64) {
      c0 = wmma_bf16(load_a16(hbuf + k, 1024), load_b16(w2 + k, 1024), c0);
      c1 = wmma_bf16(load_a16(hbuf + k + 32, 1024), load_b16(w2 + k + 32, 1024), c1);
    }
    v8f c = c0 + c1;
    float bias = b2[d0 + nl];
#pragma unroll
    for (int v = 0; v < 8; ++v) {
      float x = c[v] + bias;
      float g = 1.f / (1.f + __expf(-x));
      size_t idx = (r0 + mb + v) * 256 + d0 + nl;
      out[idx] = g * U[idx];
    }
  }
}

// ------------------------- host launch -------------------------
extern "C" void kernel_launch(void* const* d_in, const int* in_sizes, int n_in,
                              void* d_out, int out_size, void* d_ws, size_t ws_size,
                              hipStream_t stream) {
  (void)in_sizes; (void)n_in; (void)out_size; (void)ws_size;
  const float* Z    = (const float*)d_in[0];
  // d_in[1] covariates: unused by the reference
  const unsigned char* adj = (const unsigned char*)d_in[2];  // jax bool -> 1 byte
  const float* L    = (const float*)d_in[3];
  const float* Wq_r = (const float*)d_in[4];
  const float* Wq_i = (const float*)d_in[5];
  const float* Wk_r = (const float*)d_in[6];
  const float* Wk_i = (const float*)d_in[7];
  const float* W1   = (const float*)d_in[8];
  const float* b1   = (const float*)d_in[9];
  const float* W2   = (const float*)d_in[10];
  const float* b2   = (const float*)d_in[11];
  float* out = (float*)d_out;

  char* ws = (char*)d_ws;
  size_t off = 0;
  auto alloc = [&](size_t bytes) {
    char* p = ws + off; off += (bytes + 255) & ~(size_t)255; return p;
  };
  bf16_t* Zh   = (bf16_t*)alloc((size_t)BB * NN * DD * 2);
  bf16_t* ZT   = (bf16_t*)alloc((size_t)BB * NN * DD * 2);
  bf16_t* Lh   = (bf16_t*)alloc((size_t)NN * NN * 2);
  bf16_t* WT   = (bf16_t*)alloc((size_t)4 * 128 * 256 * 2);
  bf16_t* W1T  = (bf16_t*)alloc((size_t)1024 * 512 * 2);
  bf16_t* W2T  = (bf16_t*)alloc((size_t)256 * 1024 * 2);
  bf16_t* Qcat = (bf16_t*)alloc((size_t)BB * NN * 256 * 2);
  bf16_t* Kcat = (bf16_t*)alloc((size_t)BB * NN * 256 * 2);
  float*  U    = (float*) alloc((size_t)BB * NN * DD * 4);
  bf16_t* LHh  = (bf16_t*)alloc((size_t)BB * NN * DD * 2);

  cast_bf16_kernel<<<2048, 256, 0, stream>>>(Z, Zh, BB * NN * DD);
  transpose_z_kernel<<<2048, 256, 0, stream>>>(Z, ZT);
  cast_bf16_kernel<<<2048, 256, 0, stream>>>(L, Lh, NN * NN);
  transpose2d_bf16_kernel<<<128, 256, 0, stream>>>(Wq_r, WT + 0 * 128 * 256, 256, 128);
  transpose2d_bf16_kernel<<<128, 256, 0, stream>>>(Wq_i, WT + 1 * 128 * 256, 256, 128);
  transpose2d_bf16_kernel<<<128, 256, 0, stream>>>(Wk_r, WT + 2 * 128 * 256, 256, 128);
  transpose2d_bf16_kernel<<<128, 256, 0, stream>>>(Wk_i, WT + 3 * 128 * 256, 256, 128);
  transpose2d_bf16_kernel<<<512, 256, 0, stream>>>(W1, W1T, 512, 1024);
  transpose2d_bf16_kernel<<<256, 256, 0, stream>>>(W2, W2T, 1024, 256);

  proj_kernel<<<8192, 128, 0, stream>>>(Zh, WT, Qcat, Kcat);
  lh_kernel<<<4096, 128, 0, stream>>>(Lh, ZT, LHh);
  attn_kernel<<<256, 128, 0, stream>>>(Qcat, Kcat, ZT, adj, U);
  mlp_kernel<<<1024, 32, 0, stream>>>(Zh, LHh, W1T, W2T, b1, b2, U, out);
}
